// MIMS_19241453486105
// MI455X (gfx1250) — compile-verified, matrix operand order checked by reading, not run on hardware
//
#include <hip/hip_runtime.h>
#include <hip/hip_bf16.h>
#include <math.h>

// ---------------------------------------------------------------------------
// MIMS forward for MI455X (gfx1250, wave32, WMMA 16x16x32 f16)
// ---------------------------------------------------------------------------

#define L_SEQ 32768   // 32*32*32 voxels after stride-2 conv
#define NVOX  32768

typedef __attribute__((ext_vector_type(16))) _Float16 v16h;
typedef __attribute__((ext_vector_type(8)))  _Float16 v8h;
typedef __attribute__((ext_vector_type(8)))  float    v8f;
typedef __attribute__((ext_vector_type(4)))  int      v4i;

// gfx1250 async global->LDS copy path (guarded: falls back to plain loads)
#if defined(__has_builtin)
#if __has_builtin(__builtin_amdgcn_global_load_async_to_lds_b128) && \
    __has_builtin(__builtin_amdgcn_s_wait_asynccnt)
#define MIMS_ASYNC_LDS 1
#endif
#endif

__device__ __forceinline__ int perm_index(int l, int perm) {
  // perm 0: l = d*1024 + h*32 + w  (identity)
  // perm 1: l = d*1024 + w*32 + h  -> voxel = d*1024 + h*32 + w
  // perm 2: l = h*1024 + w*32 + d  -> voxel = d*1024 + h*32 + w
  if (perm == 0) return l;
  int a = l >> 10, b = (l >> 5) & 31, c = l & 31;
  if (perm == 1) return (a << 10) | (c << 5) | b;
  return (c << 10) | (a << 5) | b;
}

__device__ __forceinline__ float silu_f(float x) { return x / (1.f + __expf(-x)); }

// ---------------------------------------------------------------------------
// 1) Embedding: conv3d 7x7x7 stride2 pad3 (single in-channel per modality)
// ---------------------------------------------------------------------------
__global__ void embed_conv_kernel(const float* __restrict__ x,
                                  const float* __restrict__ ew,
                                  float* __restrict__ CONV) {
  int i = blockIdx.x * blockDim.x + threadIdx.x;       // 4*16*32768 threads
  if (i >= 4 * 16 * NVOX) return;
  int v   = i & (NVOX - 1);
  int co  = (i >> 15) & 15;
  int mod = i >> 19;
  int od = v >> 10, oh = (v >> 5) & 31, ow = v & 31;
  const float* w   = ew + ((size_t)mod * 16 + co) * 343;
  const float* xin = x + (size_t)mod * 262144;         // 64^3
  float acc = 0.f;
  for (int kd = 0; kd < 7; ++kd) {
    int id = od * 2 - 3 + kd;
    if ((unsigned)id >= 64u) continue;
    for (int kh = 0; kh < 7; ++kh) {
      int ih = oh * 2 - 3 + kh;
      if ((unsigned)ih >= 64u) continue;
      const float* xr = xin + ((size_t)id * 64 + ih) * 64;
      const float* wr = w + (kd * 7 + kh) * 7;
      for (int kw = 0; kw < 7; ++kw) {
        int iw = ow * 2 - 3 + kw;
        if ((unsigned)iw >= 64u) continue;
        acc += xr[iw] * wr[kw];
      }
    }
  }
  CONV[i] = acc;
}

// instance-norm stats per (mod,channel): 64 blocks
__global__ void inorm_stats_kernel(const float* __restrict__ CONV,
                                   float* __restrict__ stats) {
  int ch = blockIdx.x;                                  // 0..63
  const float* p = CONV + (size_t)ch * NVOX;
  float s = 0.f, s2 = 0.f;
  for (int i = threadIdx.x; i < NVOX; i += blockDim.x) {
    float v = p[i]; s += v; s2 += v * v;
  }
  __shared__ float sh[256], sh2[256];
  sh[threadIdx.x] = s; sh2[threadIdx.x] = s2;
  __syncthreads();
  for (int off = 128; off; off >>= 1) {
    if ((int)threadIdx.x < off) {
      sh[threadIdx.x]  += sh[threadIdx.x + off];
      sh2[threadIdx.x] += sh2[threadIdx.x + off];
    }
    __syncthreads();
  }
  if (threadIdx.x == 0) {
    float m = sh[0] / (float)NVOX;
    float var = sh2[0] / (float)NVOX - m * m;
    stats[ch]      = m;
    stats[64 + ch] = rsqrtf(var + 1e-5f);
  }
}

__global__ void inorm_gelu_kernel(const float* __restrict__ CONV,
                                  const float* __restrict__ stats,
                                  float* __restrict__ U) {
  int i = blockIdx.x * blockDim.x + threadIdx.x;
  if (i >= 4 * 16 * NVOX) return;
  int ch = i >> 15;                                     // 0..63
  float u = (CONV[i] - stats[ch]) * stats[64 + ch];
  U[i] = 0.5f * u * (1.f + erff(u * 0.70710678118654752f));
}

// ---------------------------------------------------------------------------
// 2) Mamba building blocks
// ---------------------------------------------------------------------------

// gather voxels in run-permuted order + channel LayerNorm.
// SEQ has a fixed row stride of 64 floats; cols [dm,64) are zero-filled so the
// WMMA GEMM can read Kpad columns unconditionally.
__global__ void gather_ln_kernel(const float* __restrict__ vol,   // (dm, NVOX)
                                 const float* __restrict__ lnw,
                                 const float* __restrict__ lnb,
                                 float* __restrict__ seq,         // (L, 64)
                                 int dm, int perm) {
  int l = blockIdx.x * blockDim.x + threadIdx.x;
  if (l >= L_SEQ) return;
  int v = perm_index(l, perm);
  float m = 0.f;
  for (int c = 0; c < dm; ++c) m += vol[(size_t)c * NVOX + v];
  m /= (float)dm;
  float var = 0.f;
  for (int c = 0; c < dm; ++c) { float d = vol[(size_t)c * NVOX + v] - m; var += d * d; }
  var /= (float)dm;
  float inv = rsqrtf(var + 1e-5f);
  float* o = seq + (size_t)l * 64;
  for (int c = 0; c < dm; ++c)
    o[c] = (vol[(size_t)c * NVOX + v] - m) * inv * lnw[c] + lnb[c];
  for (int c = dm; c < 64; ++c) o[c] = 0.f;
}

// pack an (N x K) f32 weight into zero-padded (Npad x Kpad) f16
__global__ void pack_w_kernel(const float* __restrict__ W, int N, int K,
                              _Float16* __restrict__ Wh, int Npad, int Kpad) {
  int i = blockIdx.x * blockDim.x + threadIdx.x;
  if (i >= Npad * Kpad) return;
  int n = i / Kpad, k = i - n * Kpad;
  Wh[i] = (n < N && k < K) ? (_Float16)W[(size_t)n * K + k] : (_Float16)0.f;
}

// C[m,n] = sum_k A[m,k] * Wh[n,k].
// Block = 8 waves = 8 consecutive M-tiles sharing ONE N-tile, so the 16-row
// weight panel (16 x Kpad f16, <= 4KB) is staged into LDS once per block
// (async global->LDS on gfx1250 when available) and B fragments come from LDS.
// A: f32, row stride lda (>= Kpad, zero-padded). Wh: packed f16 (Npad x Kpad).
__global__ __launch_bounds__(256)
void wmma_gemm_kernel(const float* __restrict__ A, int lda,
                      const _Float16* __restrict__ Wh, int Kpad,
                      float* __restrict__ C, int ldc,
                      int M, int N) {
  __shared__ _Float16 shw[16 * 128];          // up to Kpad=128
  int ntn = (N + 15) >> 4;
  int bt = blockIdx.x;
  int mg = bt / ntn, tn = bt - mg * ntn;
  int wv = threadIdx.x >> 5;
  int lane = threadIdx.x & 31;
  int m0 = (mg * 8 + wv) << 4;                // M == 32768, always in range
  int n0 = tn << 4;

  // ---- stage weight panel rows n0..n0+15 (contiguous 16*Kpad halves) ----
  int nch = 2 * Kpad;                          // number of 16-byte chunks
  int tid = threadIdx.x;
#ifdef MIMS_ASYNC_LDS
  if (tid < nch) {
    const _Float16* gsrc = Wh + (size_t)n0 * Kpad + tid * 8;
    __builtin_amdgcn_global_load_async_to_lds_b128(
        (__attribute__((address_space(1))) v4i*)gsrc,
        (__attribute__((address_space(3))) v4i*)&shw[tid * 8], 0, 0);
  }
  __builtin_amdgcn_s_wait_asynccnt(0);
  __syncthreads();
#else
  if (tid < nch) {
    v8h tmp = *(const v8h*)(Wh + (size_t)n0 * Kpad + tid * 8);
    *(v8h*)&shw[tid * 8] = tmp;
  }
  __syncthreads();
#endif

  int half = lane >> 4;            // which K-half this lane holds
  int l16  = lane & 15;
  const float*    Ar = A + (size_t)(m0 + l16) * lda + half * 8;
  const _Float16* Wl = shw + l16 * Kpad + half * 8;
  v8f acc = {0.f, 0.f, 0.f, 0.f, 0.f, 0.f, 0.f, 0.f};
#pragma unroll 2
  for (int k0 = 0; k0 < Kpad; k0 += 32) {
    v8f alo = *(const v8f*)(Ar + k0);         // K = kb+0..7
    v8f ahi = *(const v8f*)(Ar + k0 + 16);    // K = kb+16..23
    v8h al = __builtin_convertvector(alo, v8h);
    v8h ah = __builtin_convertvector(ahi, v8h);
    v8h bl = *(const v8h*)(Wl + k0);
    v8h bh = *(const v8h*)(Wl + k0 + 16);
    v16h a, b;
#pragma unroll
    for (int e = 0; e < 8; ++e) {
      a[e] = al[e]; a[8 + e] = ah[e];
      b[e] = bl[e]; b[8 + e] = bh[e];
    }
    acc = __builtin_amdgcn_wmma_f32_16x16x32_f16(false, a, false, b,
                                                 (short)0, acc, false, false);
  }
  // C/D layout: lane<16 -> n=lane, m=m0+v ; lane>=16 -> n=lane-16, m=m0+v+8
#pragma unroll
  for (int v = 0; v < 8; ++v) {
    int m = m0 + v + half * 8;
    int n = n0 + l16;
    if (n < N) C[(size_t)m * ldc + n] = acc[v];
  }
}

// causal depthwise conv (K=4) + SiLU; optional sequence flip on gather
__global__ void dwconv_silu_kernel(const float* __restrict__ X, int ldx,
                                   const float* __restrict__ cw,
                                   const float* __restrict__ cb,
                                   float* __restrict__ UC,
                                   int L, int di, int flip) {
  int i = blockIdx.x * blockDim.x + threadIdx.x;
  if (i >= L * di) return;
  int l = i / di, d = i - l * di;
  float acc = cb[d];
#pragma unroll
  for (int k = 0; k < 4; ++k) {
    int lk = l - 3 + k;
    if (lk >= 0) {
      int src = flip ? (L - 1 - lk) : lk;
      acc += X[(size_t)src * ldx + d] * cw[d * 4 + k];
    }
  }
  UC[(size_t)l * di + d] = silu_f(acc);
}

// dt = softplus(xdbl[:, :dr] @ dt_w.T + dt_b)
__global__ void dt_softplus_kernel(const float* __restrict__ XDBL, int ldx,
                                   const float* __restrict__ dtw,
                                   const float* __restrict__ dtb,
                                   float* __restrict__ DT,
                                   int L, int di, int dr) {
  int i = blockIdx.x * blockDim.x + threadIdx.x;
  if (i >= L * di) return;
  int l = i / di, d = i - l * di;
  float acc = dtb[d];
  for (int r = 0; r < dr; ++r) acc += XDBL[(size_t)l * ldx + r] * dtw[d * dr + r];
  DT[i] = (acc > 20.f) ? acc : log1pf(__expf(acc));
}

// selective scan: lanes 0..15 = states of channel 2*wave, lanes 16..31 = 2*wave+1
__global__ void ssm_scan_kernel(const float* __restrict__ UC,
                                const float* __restrict__ DT,
                                const float* __restrict__ XDBL, int ldx,
                                const float* __restrict__ A_log,
                                const float* __restrict__ Dp,
                                float* __restrict__ YD,
                                int L, int di, int dr, int flip, int first) {
  int wave = (blockIdx.x * blockDim.x + threadIdx.x) >> 5;
  int lane = threadIdx.x & 31;
  int d = wave * 2 + (lane >> 4);
  int s = lane & 15;
  if (d >= di) return;
  const float A  = -__expf(A_log[d * 16 + s]);
  const float Dv = Dp[d];
  float h = 0.f;
  for (int t = 0; t < L; ++t) {
    if (t + 16 < L) {
      __builtin_prefetch(&DT[(size_t)(t + 16) * di + d], 0, 1);
      __builtin_prefetch(&XDBL[(size_t)(t + 16) * ldx + dr + s], 0, 1);
    }
    float dt = DT[(size_t)t * di + d];
    float u  = UC[(size_t)t * di + d];
    float Bv = XDBL[(size_t)t * ldx + dr + s];
    float Cv = XDBL[(size_t)t * ldx + dr + 16 + s];
    h = h * __expf(dt * A) + dt * u * Bv;
    float contrib = h * Cv;
#pragma unroll
    for (int m = 1; m < 16; m <<= 1) contrib += __shfl_xor(contrib, m, 32);
    if (s == 0) {
      float y = contrib + u * Dv;
      int l = flip ? (L - 1 - t) : t;
      float* p = &YD[(size_t)l * di + d];
      if (first) *p = y; else *p += y;
    }
  }
}

// y *= silu(z), z = xz[..., di:]
__global__ void mul_silu_z_kernel(float* __restrict__ YD,
                                  const float* __restrict__ XZ, int ldxz,
                                  int di, int L) {
  int i = blockIdx.x * blockDim.x + threadIdx.x;
  if (i >= L * di) return;
  int l = i / di, d = i - l * di;
  YD[i] *= silu_f(XZ[(size_t)l * ldxz + di + d]);
}

// scatter run output back to volume layout (bijective index -> plain add)
__global__ void scatter_add_kernel(const float* __restrict__ OUTSEQ,
                                   float* __restrict__ target,
                                   int dm, int perm) {
  int i = blockIdx.x * blockDim.x + threadIdx.x;
  if (i >= L_SEQ * dm) return;
  int l = i / dm, c = i - l * dm;
  int v = perm_index(l, perm);
  target[(size_t)c * NVOX + v] += OUTSEQ[(size_t)l * dm + c];
}

// ---------------------------------------------------------------------------
// 3) tail: combine + upsample
// ---------------------------------------------------------------------------
__global__ void combine_kernel(const float* __restrict__ ACC,
                               const float* __restrict__ INTER,
                               const float* __restrict__ U,
                               const float* __restrict__ reduce_w,
                               float* __restrict__ YF) {
  int i = blockIdx.x * blockDim.x + threadIdx.x;     // 16 * NVOX
  if (i >= 16 * NVOX) return;
  int c = i >> 15, v = i & (NVOX - 1);
  float rf = 0.f;
  for (int cc = 0; cc < 64; ++cc)
    rf += reduce_w[c * 64 + cc] * INTER[(size_t)cc * NVOX + v];
  float es = U[(size_t)(c)      * NVOX + v] + U[(size_t)(16 + c) * NVOX + v] +
             U[(size_t)(32 + c) * NVOX + v] + U[(size_t)(48 + c) * NVOX + v];
  // y = intra_runs + embed_sum(residuals) + inter_feat + embed_sum
  YF[i] = ACC[i] + rf + 2.f * es;
}

__global__ void upsample_kernel(const float* __restrict__ YF,
                                const float* __restrict__ up_w,
                                float* __restrict__ out) {
  int i = blockIdx.x * blockDim.x + threadIdx.x;     // 32 * 64^3
  if (i >= 32 * 262144) return;
  int sp = i & 262143, o = i >> 18;
  int dd = sp >> 12, hh = (sp >> 6) & 63, ww = sp & 63;
  int d = dd >> 1, ii = dd & 1, h = hh >> 1, j = hh & 1, w = ww >> 1, k = ww & 1;
  int v = (d << 10) | (h << 5) | w;
  float acc = 0.f;
  for (int c = 0; c < 16; ++c)
    acc += YF[(size_t)c * NVOX + v] *
           up_w[((((size_t)c * 32 + o) * 2 + ii) * 2 + j) * 2 + k];
  out[i] = acc;
}

__global__ void fill0_kernel(float* p, int n) {
  int i = blockIdx.x * blockDim.x + threadIdx.x;
  if (i < n) p[i] = 0.f;
}
__global__ void copyf_kernel(float* dst, const float* src, int n) {
  int i = blockIdx.x * blockDim.x + threadIdx.x;
  if (i < n) dst[i] = src[i];
}

// ---------------------------------------------------------------------------
// Host orchestration
// ---------------------------------------------------------------------------
struct MambaP {
  const float *in_w, *out_w, *ln_w, *ln_b;
  const float *conv_w[2], *conv_b[2], *xproj_w[2], *dt_w[2], *dt_b[2], *A_log[2], *D[2];
  int dm, di, dr;
  // packed f16 weights for WMMA GEMMs
  _Float16 *in_wh;        int kp_in;     // (2di x kp_in)
  _Float16 *xproj_wh[2];                 // (48 x di)
  _Float16 *out_wh;                      // (dm x di)
};

static MambaP make_p(void* const* d_in, int base, int dm, int di, int dr) {
  MambaP p{};
  p.in_w  = (const float*)d_in[base + 0];
  p.out_w = (const float*)d_in[base + 1];
  p.ln_w  = (const float*)d_in[base + 2];
  p.ln_b  = (const float*)d_in[base + 3];
  for (int s = 0; s < 2; ++s) {
    int b = base + 4 + 7 * s;
    p.conv_w[s]  = (const float*)d_in[b + 0];
    p.conv_b[s]  = (const float*)d_in[b + 1];
    p.xproj_w[s] = (const float*)d_in[b + 2];
    p.dt_w[s]    = (const float*)d_in[b + 3];
    p.dt_b[s]    = (const float*)d_in[b + 4];
    p.A_log[s]   = (const float*)d_in[b + 5];
    p.D[s]       = (const float*)d_in[b + 6];
  }
  p.dm = dm; p.di = di; p.dr = dr;
  return p;
}

static inline int gb(long n, int b) { return (int)((n + b - 1) / b); }

static _Float16* pack_weight(const float* W, int N, int K, int Npad, int Kpad,
                             _Float16** pool, hipStream_t s) {
  _Float16* dst = *pool;
  *pool += (size_t)Npad * Kpad;                // all sizes multiples of 64 halves
  pack_w_kernel<<<gb((long)Npad * Kpad, 256), 256, 0, s>>>(W, N, K, dst, Npad, Kpad);
  return dst;
}

static void launch_gemm(const float* A, int lda, const _Float16* Wh, int Kpad,
                        float* C, int ldc, int M, int N, hipStream_t s) {
  int ntn = (N + 15) / 16;
  int mg  = M / 128;                 // 8 M-tiles per block; M == 32768
  wmma_gemm_kernel<<<mg * ntn, 256, 0, s>>>(A, lda, Wh, Kpad, C, ldc, M, N);
}

static void run_mamba_once(const MambaP& P, const float* vol, float* target, int perm,
                           float* SEQ, float* XZ, float* UC, float* XDBL,
                           float* DT, float* YD, float* OUTSEQ, hipStream_t stream) {
  const int L = L_SEQ;
  const int dm = P.dm, di = P.di, dr = P.dr;
  const int ldx = dr + 32;

  gather_ln_kernel<<<gb(L, 256), 256, 0, stream>>>(vol, P.ln_w, P.ln_b, SEQ, dm, perm);
  // in_proj: XZ(L x 2di) = SEQ @ in_w^T   (SEQ row stride 64, zero-padded)
  launch_gemm(SEQ, 64, P.in_wh, P.kp_in, XZ, 2 * di, L, 2 * di, stream);

  for (int dir = 0; dir < 2; ++dir) {
    dwconv_silu_kernel<<<gb((long)L * di, 256), 256, 0, stream>>>(
        XZ, 2 * di, P.conv_w[dir], P.conv_b[dir], UC, L, di, dir);
    // xproj: XDBL(L x (dr+32)) = UC @ xproj_w^T
    launch_gemm(UC, di, P.xproj_wh[dir], di, XDBL, ldx, L, dr + 32, stream);
    dt_softplus_kernel<<<gb((long)L * di, 256), 256, 0, stream>>>(
        XDBL, ldx, P.dt_w[dir], P.dt_b[dir], DT, L, di, dr);
    int waves = di / 2;
    ssm_scan_kernel<<<gb((long)waves * 32, 256), 256, 0, stream>>>(
        UC, DT, XDBL, ldx, P.A_log[dir], P.D[dir], YD, L, di, dr, dir, dir == 0);
  }

  mul_silu_z_kernel<<<gb((long)L * di, 256), 256, 0, stream>>>(YD, XZ, 2 * di, di, L);
  // out_proj: OUTSEQ(L x dm) = YD @ out_w^T
  launch_gemm(YD, di, P.out_wh, di, OUTSEQ, dm, L, dm, stream);
  scatter_add_kernel<<<gb((long)L * dm, 256), 256, 0, stream>>>(OUTSEQ, target, dm, perm);
}

extern "C" void kernel_launch(void* const* d_in, const int* in_sizes, int n_in,
                              void* d_out, int out_size, void* d_ws, size_t ws_size,
                              hipStream_t stream) {
  (void)in_sizes; (void)n_in; (void)out_size;

  const float* x       = (const float*)d_in[0];
  const float* embed_w = (const float*)d_in[1];
  MambaP intraP = make_p(d_in, 2,  16, 32,  1);
  MambaP interP = make_p(d_in, 20, 64, 128, 4);
  const float* reduce_w = (const float*)d_in[38];
  const float* up_w     = (const float*)d_in[39];
  float* out = (float*)d_out;

  // ---- workspace layout (floats; every chunk a multiple of 128 floats) ----
  float* ws = (float*)d_ws;
  size_t off = 0;
  auto alloc = [&](size_t n) { float* p = ws + off; off += n; return p; };
  float* CONV   = alloc((size_t)4 * 16 * NVOX);   // reused as SEQ (L x 64) later
  float* STATS  = alloc(128);
  float* U      = alloc((size_t)4 * 16 * NVOX);   // == u_cat (64, NVOX)
  float* ACC    = alloc((size_t)16 * NVOX);       // intra run accumulator
  float* INTER  = alloc((size_t)64 * NVOX);       // inter mamba output
  float* XZ     = alloc((size_t)L_SEQ * 256);
  float* UC     = alloc((size_t)L_SEQ * 128);
  float* XDBL   = alloc((size_t)L_SEQ * 36);
  float* DT     = alloc((size_t)L_SEQ * 128);
  float* YD     = alloc((size_t)L_SEQ * 128);
  float* OUTSEQ = alloc((size_t)L_SEQ * 64);
  float* YF     = alloc((size_t)16 * NVOX);
  float* WPOOLf = alloc(21504);                   // 43008 f16 packed weights
  if (ws_size < off * sizeof(float)) return;      // insufficient scratch
  float* SEQ = CONV;                              // recycle after embeddings

  // ---- pack WMMA weights (f32 -> zero-padded f16) ----
  _Float16* pool = (_Float16*)WPOOLf;
  intraP.kp_in = 32;
  intraP.in_wh        = pack_weight(intraP.in_w,     64, 16,  64, 32, &pool, stream);
  intraP.xproj_wh[0]  = pack_weight(intraP.xproj_w[0], 33, 32, 48, 32, &pool, stream);
  intraP.xproj_wh[1]  = pack_weight(intraP.xproj_w[1], 33, 32, 48, 32, &pool, stream);
  intraP.out_wh       = pack_weight(intraP.out_w,    16, 32,  16, 32, &pool, stream);
  interP.kp_in = 64;
  interP.in_wh        = pack_weight(interP.in_w,    256, 64, 256, 64, &pool, stream);
  interP.xproj_wh[0]  = pack_weight(interP.xproj_w[0], 36, 128, 48, 128, &pool, stream);
  interP.xproj_wh[1]  = pack_weight(interP.xproj_w[1], 36, 128, 48, 128, &pool, stream);
  interP.out_wh       = pack_weight(interP.out_w,    64, 128, 64, 128, &pool, stream);

  // ---- embeddings ----
  embed_conv_kernel<<<gb((long)4 * 16 * NVOX, 256), 256, 0, stream>>>(x, embed_w, CONV);
  inorm_stats_kernel<<<64, 256, 0, stream>>>(CONV, STATS);
  inorm_gelu_kernel<<<gb((long)4 * 16 * NVOX, 256), 256, 0, stream>>>(CONV, STATS, U);

  // ---- init accumulators ----
  fill0_kernel<<<gb((long)16 * NVOX, 256), 256, 0, stream>>>(ACC, 16 * NVOX);
  copyf_kernel<<<gb((long)64 * NVOX, 256), 256, 0, stream>>>(INTER, U, 64 * NVOX); // +u_cat residual

  // ---- intra mamba: 4 modalities x 3 axis runs ----
  for (int i = 0; i < 4; ++i)
    for (int perm = 0; perm < 3; ++perm)
      run_mamba_once(intraP, U + (size_t)i * 16 * NVOX, ACC, perm,
                     SEQ, XZ, UC, XDBL, DT, YD, OUTSEQ, stream);

  // ---- inter mamba: u_cat x 3 axis runs ----
  for (int perm = 0; perm < 3; ++perm)
    run_mamba_once(interP, U, INTER, perm,
                   SEQ, XZ, UC, XDBL, DT, YD, OUTSEQ, stream);

  // ---- combine + upsample ----
  combine_kernel<<<gb((long)16 * NVOX, 256), 256, 0, stream>>>(ACC, INTER, U, reduce_w, YF);
  upsample_kernel<<<gb((long)32 * 262144, 256), 256, 0, stream>>>(YF, up_w, out);
}
